// SparseCompressor_60576218743271
// MI455X (gfx1250) — matrix operand order repaired and checked by reading, not compiled, not run on hardware
//
#include <hip/hip_runtime.h>
#include <hip/hip_bf16.h>
#include <math.h>
#include <stdint.h>

// Problem constants (B=4, S=2048, D=1024, R=128, N=64, K=8)
#define T_  8192        // B*S tokens
#define D_  1024
#define R_  128
#define N_  64
#define K_  8
#define TK_ (T_ * K_)   // 65536 (token, k) pairs

typedef __attribute__((ext_vector_type(8)))  float  v8f;
typedef __attribute__((ext_vector_type(16))) __bf16 v16bf;

union ABfrag { v16bf v; uint4 q[2]; unsigned d[8]; };

// Split a pair of floats into packed-bf16 (hi, lo) dwords: dword = {lo16: a, hi16: b}
__device__ __forceinline__ void split_pack(float a, float b,
                                           unsigned& dhi, unsigned& dlo) {
  __bf16 ah = (__bf16)a, bh = (__bf16)b;
  float  al = a - (float)ah;
  float  bl = b - (float)bh;
  __bf16 alh = (__bf16)al, blh = (__bf16)bl;
  dhi = (unsigned)__builtin_bit_cast(unsigned short, ah) |
        ((unsigned)__builtin_bit_cast(unsigned short, bh) << 16);
  dlo = (unsigned)__builtin_bit_cast(unsigned short, alh) |
        ((unsigned)__builtin_bit_cast(unsigned short, blh) << 16);
}

// -------------------------------------------------------------------------
// Kernel 0: one-shot split of neurons into bf16 hi/lo packed-(k,k+1) planes.
// Layout: plane[n][kp][col] (dword), kp = k/2 -> B-frag-ready, copy-stageable.
// -------------------------------------------------------------------------
__global__ __launch_bounds__(256) void split_neurons_kernel(
    const float* __restrict__ neurons, unsigned* __restrict__ nh,
    unsigned* __restrict__ nl)
{
  int gid = blockIdx.x * 256 + threadIdx.x;      // over N_*512*32 = 1M units
  int c4 = gid & 31;                             // column group of 4
  int kp = (gid >> 5) & 511;                     // k-pair index
  int n  = gid >> 14;
  const float4 f0 = *(const float4*)(neurons + (size_t)n * (D_ * R_) +
                                     (size_t)(2 * kp)     * R_ + c4 * 4);
  const float4 f1 = *(const float4*)(neurons + (size_t)n * (D_ * R_) +
                                     (size_t)(2 * kp + 1) * R_ + c4 * 4);
  uint4 ph, pl;
  split_pack(f0.x, f1.x, ph.x, pl.x);
  split_pack(f0.y, f1.y, ph.y, pl.y);
  split_pack(f0.z, f1.z, ph.z, pl.z);
  split_pack(f0.w, f1.w, ph.w, pl.w);
  *(uint4*)(nh + (size_t)gid * 4) = ph;          // == n*65536 + kp*128 + c4*4
  *(uint4*)(nl + (size_t)gid * 4) = pl;
}

// -------------------------------------------------------------------------
// Kernel 1: router scores + top-8 + softmax + expert counting
// -------------------------------------------------------------------------
__global__ __launch_bounds__(256) void router_kernel(
    const float* __restrict__ x, const float* __restrict__ rw,
    float* __restrict__ outW, int* __restrict__ outI, int* __restrict__ counts)
{
  __shared__ float xs[16 * 1024];
  __shared__ float rch[64 * 65];
  __shared__ float sc[16 * 64];

  const int tid = threadIdx.x;
  const int t0  = blockIdx.x * 16;

  {
    const float4* xg = (const float4*)(x + (size_t)t0 * D_);
    float4* xs4 = (float4*)xs;
#pragma unroll
    for (int i = 0; i < 16; ++i) xs4[tid + i * 256] = xg[tid + i * 256];
  }

  const int e = tid & 63;
  const int g = tid >> 6;
  float a0 = 0.f, a1 = 0.f, a2 = 0.f, a3 = 0.f;
  const float4* rw4 = (const float4*)rw;

  for (int c = 0; c < 16; ++c) {
    __syncthreads();
#pragma unroll
    for (int q = 0; q < 4; ++q) {
      int f  = tid * 4 + q;
      int ee = f >> 4, jj = f & 15;
      float4 v = rw4[ee * 256 + c * 16 + jj];
      rch[ee * 65 + jj * 4 + 0] = v.x;
      rch[ee * 65 + jj * 4 + 1] = v.y;
      rch[ee * 65 + jj * 4 + 2] = v.z;
      rch[ee * 65 + jj * 4 + 3] = v.w;
    }
    __syncthreads();
    const float* xb = xs + c * 64;
#pragma unroll 8
    for (int j = 0; j < 64; ++j) {
      float rv = rch[e * 65 + j];
      a0 += xb[(g * 4 + 0) * 1024 + j] * rv;
      a1 += xb[(g * 4 + 1) * 1024 + j] * rv;
      a2 += xb[(g * 4 + 2) * 1024 + j] * rv;
      a3 += xb[(g * 4 + 3) * 1024 + j] * rv;
    }
  }
  sc[(g * 4 + 0) * 64 + e] = a0;
  sc[(g * 4 + 1) * 64 + e] = a1;
  sc[(g * 4 + 2) * 64 + e] = a2;
  sc[(g * 4 + 3) * 64 + e] = a3;
  __syncthreads();

  if (tid < 16) {
    const int t = t0 + tid;
    float topv[K_]; int topi[K_];
#pragma unroll
    for (int k = 0; k < K_; ++k) {
      float m = -3.402823466e38f; int mi = 0;
      for (int ee = 0; ee < N_; ++ee) {
        float v = sc[tid * 64 + ee];
        if (v > m) { m = v; mi = ee; }
      }
      topv[k] = m; topi[k] = mi;
      sc[tid * 64 + mi] = -3.402823466e38f;
    }
    float mx = topv[0], sum = 0.f, wv[K_];
#pragma unroll
    for (int k = 0; k < K_; ++k) { wv[k] = expf(topv[k] - mx); sum += wv[k]; }
    float inv = 1.f / sum;
#pragma unroll
    for (int k = 0; k < K_; ++k) {
      outW[t * K_ + k] = wv[k] * inv;
      outI[t * K_ + k] = topi[k];
      atomicAdd(&counts[topi[k]], 1);
    }
  }
}

// -------------------------------------------------------------------------
// Kernel 2: exclusive scan of 64 expert counts
// -------------------------------------------------------------------------
__global__ void scan_kernel(const int* __restrict__ counts,
                            int* __restrict__ offsets, int* __restrict__ cursors)
{
  if (threadIdx.x == 0) {
    int run = 0;
    for (int n = 0; n < N_; ++n) {
      offsets[n] = run; cursors[n] = run; run += counts[n];
    }
  }
}

// -------------------------------------------------------------------------
// Kernel 3: scatter (t,k) pairs into per-expert segments
// -------------------------------------------------------------------------
__global__ __launch_bounds__(256) void scatter_kernel(
    const int* __restrict__ outI, int* __restrict__ cursors,
    int* __restrict__ pair_slot)
{
  int t = blockIdx.x * 256 + threadIdx.x;
#pragma unroll
  for (int k = 0; k < K_; ++k) {
    int n = outI[t * K_ + k];
    int pos = atomicAdd(&cursors[n], 1);
    pair_slot[pos] = t * K_ + k;
  }
}

// -------------------------------------------------------------------------
// Kernel 4: grouped GEMM per expert, split-bf16 V_WMMA_F32_16X16X32_BF16.
// Weight slabs are pre-split bf16 pairs -> staged into double-buffered LDS
// with GLOBAL_LOAD_ASYNC_TO_LDS_B128 (ASYNCcnt) overlapping compute.
// grid = (512 tiles, 64 experts), block = 256 threads (8 waves).
// -------------------------------------------------------------------------
#define XSP 520   // xs pitch in dwords (512 + 8 pad: conflict-free b128 frags)
#define WSP 132   // w slab pitch in dwords (128 + 4 pad, 16B aligned)

__global__ __launch_bounds__(256) void gemm_kernel(
    const float* __restrict__ x,
    const unsigned* __restrict__ nh, const unsigned* __restrict__ nl,
    const int* __restrict__ counts, const int* __restrict__ offsets,
    const int* __restrict__ pair_slot, float* __restrict__ proj)
{
  __shared__ unsigned xs_hi[16 * XSP];        // 33.3KB bf16 pairs (d even, d odd)
  __shared__ unsigned xs_lo[16 * XSP];        // 33.3KB
  __shared__ unsigned wbuf[2][2][32 * WSP];   // 67.6KB double-buffered hi/lo slabs
  __shared__ int      slots[16];
  __shared__ int      toks[16];

  const int n   = blockIdx.y;
  const int cnt = counts[n];
  const int tb  = blockIdx.x * 16;
  if (tb >= cnt) return;
  const int base = offsets[n] + tb;
  const int rows = min(16, cnt - tb);
  const int tid  = threadIdx.x;

  if (tid < 16) {
    int s = (tid < rows) ? pair_slot[base + tid] : -1;
    slots[tid] = s;
    toks[tid]  = (s >= 0) ? (s >> 3) : -1;
  }

  const unsigned* srcH = nh + (size_t)n * (512 * 128);  // packed k-pair plane
  const unsigned* srcL = nl + (size_t)n * (512 * 128);

  // Issue async copies of one 64-k slab (32 k-pairs x 128 cols, hi+lo) into LDS
  auto issue_slab = [&](int kk, int buf) {
    const int kpb = kk >> 1;                  // first k-pair of slab
#pragma unroll
    for (int i2 = 0; i2 < 4; ++i2) {
      int u  = tid + 256 * i2;                // 1024 b128 units per plane
      int kp = u >> 5, c4 = u & 31;
      unsigned dh = (unsigned)(uintptr_t)&wbuf[buf][0][kp * WSP + c4 * 4];
      unsigned dl = (unsigned)(uintptr_t)&wbuf[buf][1][kp * WSP + c4 * 4];
      const unsigned* gh = srcH + (size_t)(kpb + kp) * 128 + c4 * 4;
      const unsigned* gl = srcL + (size_t)(kpb + kp) * 128 + c4 * 4;
      asm volatile("global_load_async_to_lds_b128 %0, %1, off"
                   :: "v"(dh), "v"(gh) : "memory");
      asm volatile("global_load_async_to_lds_b128 %0, %1, off"
                   :: "v"(dl), "v"(gl) : "memory");
    }
  };

  issue_slab(0, 0);                           // prefetch slab 0 via TDM-style async
  __syncthreads();                            // toks[] visible for the gather below

  // Gather + split 16 x rows into bf16 hi/lo LDS planes (zero-fill pad rows)
  for (int r = 0; r < 16; ++r) {
    int tk = toks[r];
    float4 v = make_float4(0.f, 0.f, 0.f, 0.f);
    if (tk >= 0) v = ((const float4*)(x + (size_t)tk * D_))[tid];
    unsigned h0, l0, h1, l1;
    split_pack(v.x, v.y, h0, l0);
    split_pack(v.z, v.w, h1, l1);
    *(uint2*)&xs_hi[r * XSP + tid * 2] = make_uint2(h0, h1);
    *(uint2*)&xs_lo[r * XSP + tid * 2] = make_uint2(l0, l1);
  }

  const int w    = tid >> 5;                  // wave id 0..7 -> R columns w*16..
  const int lane = tid & 31;
  const int h    = lane >> 4;                 // lane-half (K split per ISA layout)
  const int m    = lane & 15;                 // M (A) / N (B,C) index
  const int col  = w * 16 + m;

  v8f acc = {0.f, 0.f, 0.f, 0.f, 0.f, 0.f, 0.f, 0.f};

  asm volatile("s_wait_asynccnt 0x0" ::: "memory");
  __syncthreads();                            // slab 0 + xs ready for all waves

  for (int i = 0; i < 16; ++i) {              // 16 slabs of 64 k-values
    const int cur = i & 1;
    if (i + 1 < 16) issue_slab((i + 1) * 64, 1 - cur);  // overlap next copy

    const unsigned* wh = &wbuf[cur][0][0];
    const unsigned* wl = &wbuf[cur][1][0];
#pragma unroll
    for (int s = 0; s < 2; ++s) {             // two K=32 steps per slab
      const int kb = i * 64 + s * 32;
      ABfrag ahi, alo, bhi, blo;
      // A frag (16x32 bf16): lane half h holds K = 8h+{0..7} and 16+8h+{0..7}
      const int dbase = m * XSP + (kb >> 1);
      ahi.q[0] = *(const uint4*)&xs_hi[dbase + 4 * h];
      ahi.q[1] = *(const uint4*)&xs_hi[dbase + 8 + 4 * h];
      alo.q[0] = *(const uint4*)&xs_lo[dbase + 4 * h];
      alo.q[1] = *(const uint4*)&xs_lo[dbase + 8 + 4 * h];
      // B frag (32x16 bf16): VGPR j = pair (K=8h+2j, +1) for j<4, (K=16+8h+2(j-4), +1) for j>=4 at column col
#pragma unroll
      for (int j = 0; j < 8; ++j) {
        const int rq = s * 16 + 4 * h + j + (j & 4);
        bhi.d[j] = wh[rq * WSP + col];
        blo.d[j] = wl[rq * WSP + col];
      }
      acc = __builtin_amdgcn_wmma_f32_16x16x32_bf16(
          false, ahi.v, false, bhi.v, (short)0, acc, false, false);
      acc = __builtin_amdgcn_wmma_f32_16x16x32_bf16(
          false, ahi.v, false, blo.v, (short)0, acc, false, false);
      acc = __builtin_amdgcn_wmma_f32_16x16x32_bf16(
          false, alo.v, false, bhi.v, (short)0, acc, false, false);
    }
    if (i + 1 < 16) asm volatile("s_wait_asynccnt 0x0" ::: "memory");
    __syncthreads();                          // next slab landed; my reads done
  }

  // C/D layout: VGPR e -> M=e (lanes 0-15), M=e+8 (lanes 16-31)
#pragma unroll
  for (int e8 = 0; e8 < 8; ++e8) {
    int row = e8 + 8 * h;
    if (row < rows) {
      int slot = slots[row];
      proj[(size_t)slot * R_ + col] = acc[e8];
    }
  }
}

// -------------------------------------------------------------------------
// Kernel 5: weighted reduction over K (fixed order -> deterministic)
// -------------------------------------------------------------------------
__global__ __launch_bounds__(256) void finalize_kernel(
    const float* __restrict__ proj, const float* __restrict__ outW,
    float* __restrict__ outY)
{
  int gid = blockIdx.x * 256 + threadIdx.x;
  int t = gid >> 7, r = gid & 127;
  float acc = 0.f;
#pragma unroll
  for (int k = 0; k < K_; ++k)
    acc += outW[t * K_ + k] * proj[(size_t)(t * K_ + k) * R_ + r];
  outY[gid] = acc;
}

// -------------------------------------------------------------------------
// Launch. d_out = [output T*R | weights T*K | topk_idx T*K (int32)]
// Workspace: proj 32MB | nh 16MB | nl 16MB | counts/offsets/cursors | pairs
// -------------------------------------------------------------------------
extern "C" void kernel_launch(void* const* d_in, const int* in_sizes, int n_in,
                              void* d_out, int out_size, void* d_ws, size_t ws_size,
                              hipStream_t stream) {
  const float* x       = (const float*)d_in[0];   // [T, D]
  const float* rw      = (const float*)d_in[1];   // [N, D]
  const float* neurons = (const float*)d_in[2];   // [N, D, R]

  float* outY = (float*)d_out;                    // [T, R]
  float* outW = outY + (size_t)T_ * R_;           // [T, K]
  int*   outI = (int*)(outW + (size_t)T_ * K_);   // [T, K] int32

  char*  ws   = (char*)d_ws;
  float* proj = (float*)ws;                                   // TK_ * R_ floats
  const size_t PROJ_BYTES = (size_t)TK_ * R_ * sizeof(float); // 32MB
  const size_t PLANE_BYTES = (size_t)N_ * 512 * 128 * 4;      // 16MB each
  unsigned* nh = (unsigned*)(ws + PROJ_BYTES);
  unsigned* nl = (unsigned*)(ws + PROJ_BYTES + PLANE_BYTES);
  int* counts    = (int*)(ws + PROJ_BYTES + 2 * PLANE_BYTES);
  int* offsets   = counts + 64;
  int* cursors   = counts + 128;
  int* pair_slot = counts + 256;               // TK_ ints

  hipMemsetAsync(counts, 0, 64 * sizeof(int), stream);

  split_neurons_kernel<<<(N_ * 512 * 32) / 256, 256, 0, stream>>>(neurons, nh, nl);
  router_kernel<<<T_ / 16, 256, 0, stream>>>(x, rw, outW, outI, counts);
  scan_kernel<<<1, 64, 0, stream>>>(counts, offsets, cursors);
  scatter_kernel<<<T_ / 256, 256, 0, stream>>>(outI, cursors, pair_slot);

  dim3 ggrid(512, N_);  // covers worst-case 8192 tokens on one expert
  gemm_kernel<<<ggrid, 256, 0, stream>>>(x, nh, nl, counts, offsets, pair_slot, proj);

  finalize_kernel<<<(T_ * R_) / 256, 256, 0, stream>>>(proj, outW, outY);
}
